// PoolingAttention_80307298501264
// MI455X (gfx1250) — matrix operand order repaired and checked by reading, not compiled, not run on hardware
//
#include <hip/hip_runtime.h>

// ---------------------------------------------------------------------------
// PoolingAttention fused kernel for gfx1250 (MI455X, wave32, WMMA).
// One workgroup (256 thr = 8 wave32) per batch element; all intermediates in
// LDS; math via v_wmma_f32_16x16x32_bf16; weights pre-converted to bf16 in ws.
// ---------------------------------------------------------------------------

#define DIMC     512
#define NTOK     16
#define MPAD     32          // pools rows padded 28 -> 32
#define SCALE_F  0.125f      // 64^-0.5

typedef unsigned short u16;
typedef __attribute__((ext_vector_type(8)))  u16    u16x8;
typedef __attribute__((ext_vector_type(16))) u16    u16x16;
typedef __attribute__((ext_vector_type(16))) __bf16 bf16x16;
typedef __attribute__((ext_vector_type(8)))  float  f32x8;

// avgpool pair tables derived from GROUP1 / GROUP2
__device__ __constant__ int G1A[8] = {2, 5, 1, 0, 8, 14, 11, 10};
__device__ __constant__ int G1B[8] = {3, 6, 4, 7, 9, 15, 12, 13};
__device__ __constant__ int G2A[4] = {0, 2, 5, 4};
__device__ __constant__ int G2B[4] = {1, 3, 6, 7};

__device__ __forceinline__ u16 f2bf(float f) {            // RNE fp32 -> bf16
  unsigned u = __builtin_bit_cast(unsigned, f);
  u += 0x7FFFu + ((u >> 16) & 1u);
  return (u16)(u >> 16);
}

// A-fragment (16xK row-major, 16-bit): lane half 0 -> K[kk+0..7 | kk+16..23],
// lane half 1 -> K[kk+8..15 | kk+24..31]   (ISA 7.12.2 layout)
__device__ __forceinline__ bf16x16 frag_a(const u16* p, int ld, int kk) {
  const int lane = threadIdx.x & 31;
  const u16* r = p + (lane & 15) * ld + kk + (lane >> 4) * 8;
  u16x8 lo = *(const u16x8*)(r);
  u16x8 hi = *(const u16x8*)(r + 16);
  u16x16 v = __builtin_shufflevector(lo, hi, 0,1,2,3,4,5,6,7,8,9,10,11,12,13,14,15);
  return __builtin_bit_cast(bf16x16, v);
}

// B-fragment from B^T stored row-major (N x K): lane holds column n = lane&15,
// lane half 0 -> K[kk+0..7 | kk+16..23], lane half 1 -> K[kk+8..15 | kk+24..31]
__device__ __forceinline__ bf16x16 frag_b(const u16* p, int ld, int kk) {
  const int lane = threadIdx.x & 31;
  const u16* r = p + (lane & 15) * ld + kk + (lane >> 4) * 8;
  u16x8 lo = *(const u16x8*)(r);
  u16x8 hi = *(const u16x8*)(r + 16);
  u16x16 v = __builtin_shufflevector(lo, hi, 0,1,2,3,4,5,6,7,8,9,10,11,12,13,14,15);
  return __builtin_bit_cast(bf16x16, v);
}

__device__ __forceinline__ f32x8 wmma_bf16(bf16x16 a, bf16x16 b, f32x8 c) {
  return __builtin_amdgcn_wmma_f32_16x16x32_bf16(false, a, false, b, (short)0, c,
                                                 false, false);
}

// ---------------------------------------------------------------------------
// Weight pre-conversion: fp32 -> bf16 into workspace (wq | wkv | wproj), 2 MB.
// Runs once per launch; weights then streamed bf16 from L2 by all 4096 blocks.
// ---------------------------------------------------------------------------
__global__ void wconv_kernel(const float* __restrict__ wq,
                             const float* __restrict__ wkv,
                             const float* __restrict__ wp,
                             u16* __restrict__ out) {
  int i = blockIdx.x * blockDim.x + threadIdx.x;   // 0 .. 1048575
  float v;
  if (i < 262144)       v = wq[i];
  else if (i < 786432)  v = wkv[i - 262144];
  else                  v = wp[i - 786432];
  out[i] = f2bf(v);
}

// ---------------------------------------------------------------------------
// Fused pooling attention. Dynamic LDS layout (u16 offsets):
//   xbf   [0,      8192)   raw x bf16 (16x512)      -> reused as obf later
//   pools [8192,  24576)   LN'd pools bf16 (32x512, rows 28..31 zero)
//   qbf   [24576, 32768)   q bf16 (16x512)
//   kbf   [32768, 49152)   k bf16 per head (8 x 32x64)
//   vtb   [49152, 65536)   v^T bf16 per head (8 x 64x32)
//   pbf   [65536, 69632)   softmax probs bf16 (8 x 16x32)
//   sbuf  f32 @ byte 139264 (8 x 16x32)   -> total 155648 B
// ---------------------------------------------------------------------------
#define SMEM_BYTES 155648

__global__ __launch_bounds__(256, 1)
void pool_attn_kernel(const float* __restrict__ x,
                      const u16*   __restrict__ wbf,
                      const float* __restrict__ bproj,
                      const float* __restrict__ gamma,
                      const float* __restrict__ beta,
                      float* __restrict__ out) {
  extern __shared__ __align__(16) char smem_raw[];
  u16*   xbf   = (u16*)smem_raw;
  u16*   pools = xbf   + 8192;
  u16*   qbf   = pools + 16384;
  u16*   kbf   = qbf   + 8192;
  u16*   vtb   = kbf   + 16384;
  u16*   pbf   = vtb   + 16384;
  float* sbuf  = (float*)(pbf + 4096);
  u16*   obf   = xbf;                       // xbf dead after q-GEMM

  const u16* wq  = wbf;
  const u16* wkv = wbf + 262144;
  const u16* wpj = wbf + 786432;

  const int tid  = threadIdx.x;
  const int w    = tid >> 5;                // wave id 0..7
  const int lane = tid & 31;
  const int cn   = lane & 15;               // C-frag column within tile
  const int ch   = lane >> 4;               // C-frag row-half
  const size_t blk = blockIdx.x;
  const float* xb  = x + blk * (NTOK * DIMC);

  // ---- Phase 1: load x, build pooled rows, LayerNorm -> pools (bf16) ------
  // wave w owns rows {w, w+8, w+16, w+24}; lane owns cols lane + 32*i.
  for (int rr = w; rr < MPAD; rr += 8) {
    if (rr >= 28) {                          // zero-pad rows 28..31
      for (int i = 0; i < 16; ++i) pools[rr * DIMC + lane + 32 * i] = 0;
      continue;
    }
    float vals[16];
    float s1 = 0.f, s2 = 0.f;
    for (int i = 0; i < 16; ++i) {
      int c = lane + 32 * i;
      float v;
      if (rr < 16) {
        v = xb[rr * DIMC + c];
        xbf[rr * DIMC + c] = f2bf(v);        // raw x for q-GEMM
      } else if (rr < 24) {
        int j = rr - 16;
        v = 0.5f * (xb[G1A[j] * DIMC + c] + xb[G1B[j] * DIMC + c]);
      } else {
        int j = rr - 24;
        v = 0.5f * (xb[G2A[j] * DIMC + c] + xb[G2B[j] * DIMC + c]);
      }
      vals[i] = v; s1 += v; s2 += v * v;
    }
    for (int o = 16; o >= 1; o >>= 1) {      // wave32 reduction
      s1 += __shfl_xor(s1, o);
      s2 += __shfl_xor(s2, o);
    }
    float mu  = s1 * (1.0f / 512.0f);
    float var = s2 * (1.0f / 512.0f) - mu * mu;
    float rs  = rsqrtf(var + 1e-5f);
    for (int i = 0; i < 16; ++i) {
      int c = lane + 32 * i;
      pools[rr * DIMC + c] = f2bf((vals[i] - mu) * rs * gamma[c] + beta[c]);
    }
  }
  __syncthreads();

  // ---- Phase 2: q = x @ w_q^T  (32 n-tiles x 16 k-steps) ------------------
  for (int nt = w; nt < 32; nt += 8) {
    f32x8 acc = {0.f,0.f,0.f,0.f,0.f,0.f,0.f,0.f};
    const u16* bw = wq + nt * 16 * DIMC;     // w_q rows == B^T rows
    for (int kk = 0; kk < DIMC; kk += 32)
      acc = wmma_bf16(frag_a(xbf, DIMC, kk), frag_b(bw, DIMC, kk), acc);
    for (int r = 0; r < 8; ++r)
      qbf[(r + 8 * ch) * DIMC + nt * 16 + cn] = f2bf(acc[r]);
  }
  __syncthreads();

  // ---- Phase 3: kv = pools @ w_kv^T (2 m-tiles x 64 n-tiles) --------------
  // k stored (head,m,d); v stored transposed (head,d,m) for the PV GEMM.
  for (int j = w; j < 128; j += 8) {
    int mt = j & 1, nt = j >> 1;
    f32x8 acc = {0.f,0.f,0.f,0.f,0.f,0.f,0.f,0.f};
    const u16* ap = pools + mt * 16 * DIMC;
    const u16* bw = wkv + nt * 16 * DIMC;
    for (int kk = 0; kk < DIMC; kk += 32)
      acc = wmma_bf16(frag_a(ap, DIMC, kk), frag_b(bw, DIMC, kk), acc);
    int f = nt * 16 + cn;                    // feature 0..1023
    for (int r = 0; r < 8; ++r) {
      int m = mt * 16 + r + 8 * ch;
      u16 bv = f2bf(acc[r]);
      if (f < 512) kbf[(f >> 6) * 2048 + m * 64 + (f & 63)] = bv;
      else {
        int f2 = f - 512;
        vtb[(f2 >> 6) * 2048 + (f2 & 63) * 32 + m] = bv;
      }
    }
  }
  __syncthreads();

  // ---- Phase 4: S = q_h @ k_h^T  (8 heads x 2 key-tiles, K=64) ------------
  for (int j = w; j < 16; j += 8) {
    int h = j >> 1, nt = j & 1;
    f32x8 acc = {0.f,0.f,0.f,0.f,0.f,0.f,0.f,0.f};
    for (int kk = 0; kk < 64; kk += 32)
      acc = wmma_bf16(frag_a(qbf + h * 64, DIMC, kk),
                      frag_b(kbf + h * 2048 + nt * 16 * 64, 64, kk), acc);
    for (int r = 0; r < 8; ++r)
      sbuf[h * 512 + (r + 8 * ch) * 32 + nt * 16 + cn] = acc[r];
  }
  __syncthreads();

  // ---- Phase 5: softmax over 28 keys (one row per thread, 128 rows) -------
  if (tid < 128) {
    const float* row = sbuf + (tid >> 4) * 512 + (tid & 15) * 32;
    float mx = -1e30f;
    for (int i = 0; i < 28; ++i) { float v = row[i] * SCALE_F; mx = v > mx ? v : mx; }
    float sum = 0.f;
    for (int i = 0; i < 28; ++i) sum += __expf(row[i] * SCALE_F - mx);
    float inv = 1.0f / sum;
    u16* prow = pbf + (tid >> 4) * 512 + (tid & 15) * 32;
    for (int i = 0; i < 28; ++i) prow[i] = f2bf(__expf(row[i] * SCALE_F - mx) * inv);
    for (int i = 28; i < 32; ++i) prow[i] = 0;   // padded key cols
  }
  __syncthreads();

  // ---- Phase 6: O = P @ v  (8 heads x 4 d-tiles, single K=32 step) --------
  for (int j = w; j < 32; j += 8) {
    int h = j >> 2, nt = j & 3;
    f32x8 acc = {0.f,0.f,0.f,0.f,0.f,0.f,0.f,0.f};
    acc = wmma_bf16(frag_a(pbf + h * 512, 32, 0),
                    frag_b(vtb + h * 2048 + nt * 16 * 32, 32, 0), acc);
    for (int r = 0; r < 8; ++r)
      obf[(r + 8 * ch) * DIMC + h * 64 + nt * 16 + cn] = f2bf(acc[r]);
  }
  __syncthreads();

  // ---- Phase 7: y = O @ w_proj^T + b_proj -> global -----------------------
  for (int nt = w; nt < 32; nt += 8) {
    f32x8 acc = {0.f,0.f,0.f,0.f,0.f,0.f,0.f,0.f};
    const u16* bw = wpj + nt * 16 * DIMC;
    for (int kk = 0; kk < DIMC; kk += 32)
      acc = wmma_bf16(frag_a(obf, DIMC, kk), frag_b(bw, DIMC, kk), acc);
    float bias = bproj[nt * 16 + cn];
    for (int r = 0; r < 8; ++r)
      out[(blk * NTOK + (size_t)(r + 8 * ch)) * DIMC + nt * 16 + cn] = acc[r] + bias;
  }
}

// ---------------------------------------------------------------------------
extern "C" void kernel_launch(void* const* d_in, const int* in_sizes, int n_in,
                              void* d_out, int out_size, void* d_ws, size_t ws_size,
                              hipStream_t stream) {
  (void)n_in; (void)out_size; (void)ws_size;
  const float* x      = (const float*)d_in[0];
  const float* w_q    = (const float*)d_in[1];
  const float* w_kv   = (const float*)d_in[2];
  const float* w_proj = (const float*)d_in[3];
  const float* b_proj = (const float*)d_in[4];
  const float* gamma  = (const float*)d_in[5];
  const float* beta   = (const float*)d_in[6];
  u16* wbf = (u16*)d_ws;                        // 2 MB bf16 weights

  const int B = in_sizes[0] / (NTOK * DIMC);    // 4096

  wconv_kernel<<<1048576 / 256, 256, 0, stream>>>(w_q, w_kv, w_proj, wbf);
  pool_attn_kernel<<<B, 256, SMEM_BYTES, stream>>>(x, wbf, b_proj, gamma, beta,
                                                   (float*)d_out);
}
